// LayerGIN_MoE_48189533061200
// MI455X (gfx1250) — matrix operand order repaired and checked by reading, not compiled, not run on hardware
//
#include <hip/hip_runtime.h>
#include <hip/hip_bf16.h>

// ---------------- types for CDNA5 WMMA ----------------
typedef __attribute__((ext_vector_type(16))) __bf16 v16bf;
typedef __attribute__((ext_vector_type(8)))  __bf16 bf16x8;
typedef __attribute__((ext_vector_type(8)))  float  v8f;

static __device__ __forceinline__ v16bf cat16(bf16x8 lo, bf16x8 hi) {
    return __builtin_shufflevector(lo, hi, 0,1,2,3,4,5,6,7,8,9,10,11,12,13,14,15);
}

// ---------------- problem constants ----------------
#define NN   51200
#define CC   200
#define HH   128
#define EE   8
#define CPAD 224   // C padded to multiple of 32 for bf16 weight tiles

// ---------------- workspace layout (bytes, all 256-aligned) ----------------
static constexpr size_t O_AGG  = 0;                               // N*C f32
static constexpr size_t O_SUM1 = O_AGG  + (size_t)NN*CC*4;        // E*H f32
static constexpr size_t O_SQ1  = O_SUM1 + 4096;
static constexpr size_t O_SUM2 = O_SQ1  + 4096;
static constexpr size_t O_SQ2  = O_SUM2 + 4096;
static constexpr size_t O_IMP  = O_SQ2  + 4096;                   // 8 f32
static constexpr size_t O_LOAD = O_IMP  + 256;                    // 8 f32
static constexpr size_t O_TOP  = O_LOAD + 256;                    // 1 f32
static constexpr size_t O_CNT  = O_TOP  + 256;                    // 8 int
static constexpr size_t ZERO_END = O_CNT + 256;                   // zero [0, ZERO_END)
static constexpr size_t O_GVAL = ZERO_END;                        // N f32
static constexpr size_t O_LIST = O_GVAL + (size_t)NN*4;           // E*N int
static constexpr size_t O_SC1  = O_LIST + (size_t)EE*NN*4;        // E*H f32
static constexpr size_t O_SH1  = O_SC1  + 4096;
static constexpr size_t O_SC2  = O_SH1  + 4096;
static constexpr size_t O_SH2  = O_SC2  + 4096;
static constexpr size_t O_W1T  = O_SH2  + 4096;                   // E*H*CPAD bf16
static constexpr size_t O_W2T  = O_W1T  + (size_t)EE*HH*CPAD*2;   // E*H*H bf16
static constexpr size_t O_H1   = O_W2T  + (size_t)EE*HH*HH*2;     // E*N*H bf16

// ================= zero fill =================
__global__ __launch_bounds__(256) void zero_kernel(float4* p, int n4) {
    int i = blockIdx.x * 256 + threadIdx.x;
    if (i < n4) { float4 z = {0.f, 0.f, 0.f, 0.f}; p[i] = z; }
}

// ================= weight prep: f32 -> bf16, transposed, K-padded =================
__global__ __launch_bounds__(256) void prep_weights(const float* __restrict__ W1,
                                                    const float* __restrict__ W2,
                                                    __bf16* __restrict__ w1t,
                                                    __bf16* __restrict__ w2t) {
    int idx = blockIdx.x * 256 + threadIdx.x;
    if (idx < EE * HH * CPAD) {            // w1t[e][h][c] = W1[e][c][h], pad c>=200 -> 0
        int e = idx / (HH * CPAD);
        int r = idx % (HH * CPAD);
        int h = r / CPAD;
        int c = r % CPAD;
        float v = (c < CC) ? W1[((size_t)e * CC + c) * HH + h] : 0.f;
        w1t[idx] = (__bf16)v;
    } else {
        int i2 = idx - EE * HH * CPAD;
        if (i2 < EE * HH * HH) {           // w2t[e][h][k] = W2[e][k][h]
            int e = i2 / (HH * HH);
            int r = i2 % (HH * HH);
            int h = r / HH;
            int k = r % HH;
            w2t[i2] = (__bf16)W2[((size_t)e * HH + k) * HH + h];
        }
    }
}

// ================= gating: logits, softmax, top-1, loss accumulators, compaction =================
__global__ __launch_bounds__(256) void gating_kernel(const float* __restrict__ v,
                                                     const float* __restrict__ gW,
                                                     const float* __restrict__ gb,
                                                     float* __restrict__ gval,
                                                     float* __restrict__ impS,
                                                     float* __restrict__ loadS,
                                                     float* __restrict__ topS,
                                                     int* __restrict__ cnt,
                                                     int* __restrict__ list) {
    __shared__ float simp[EE], sload[EE], stop;
    int t = threadIdx.x;
    int n = blockIdx.x * 256 + t;
    if (t < EE) { simp[t] = 0.f; sload[t] = 0.f; }
    if (t == 0) stop = 0.f;
    __syncthreads();

    float acc[EE];
#pragma unroll
    for (int e = 0; e < EE; ++e) acc[e] = gb[e];
    const float* vr = v + (size_t)n * CC;
    for (int c = 0; c < CC; ++c) {
        float xv = vr[c];
        const float* gr = gW + c * EE;
#pragma unroll
        for (int e = 0; e < EE; ++e) acc[e] = fmaf(xv, gr[e], acc[e]);
    }
    float m = acc[0]; int idx = 0;
#pragma unroll
    for (int e = 1; e < EE; ++e) if (acc[e] > m) { m = acc[e]; idx = e; }
    float s = 0.f, ex[EE];
#pragma unroll
    for (int e = 0; e < EE; ++e) { ex[e] = expf(acc[e] - m); s += ex[e]; }
    float inv = 1.0f / s;
    float tv = inv;                         // top prob = exp(0)/s
#pragma unroll
    for (int e = 0; e < EE; ++e) atomicAdd(&simp[e], ex[e] * inv);
    atomicAdd(&sload[idx], 1.0f);
    atomicAdd(&stop, tv);
    gval[n] = tv;
    int pos = atomicAdd(&cnt[idx], 1);
    list[idx * NN + pos] = n;
    __syncthreads();
    if (t < EE) { atomicAdd(&impS[t], simp[t]); atomicAdd(&loadS[t], sload[t]); }
    if (t == 0) atomicAdd(topS, stop);
}

// ================= sparse aggregation: agg[row] += val * v[col]  (wave per edge) =================
__global__ __launch_bounds__(256) void aggregate_kernel(const int* __restrict__ a_rows,
                                                        const int* __restrict__ a_cols,
                                                        const float* __restrict__ a_vals,
                                                        const float* __restrict__ v,
                                                        float* __restrict__ agg,
                                                        int nedge) {
    int gid  = blockIdx.x * 8 + (threadIdx.x >> 5);
    int lane = threadIdx.x & 31;
    if (gid >= nedge) return;
    int row = a_rows[gid], col = a_cols[gid];
    float w = a_vals[gid];
    const float* src = v + (size_t)col * CC;
    float* dst = agg + (size_t)row * CC;
    for (int c = lane; c < CC; c += 32) atomicAdd(&dst[c], w * src[c]);
}

// ================= GEMM1: h1[e] = (agg + (1+eps)v) @ W1[e] + b1 ; store bf16, accumulate BN stats =================
__global__ __launch_bounds__(256) void gemm1_kernel(const float* __restrict__ agg,
                                                    const float* __restrict__ v,
                                                    const float* __restrict__ eps,
                                                    const __bf16* __restrict__ w1t,
                                                    const float* __restrict__ b1,
                                                    __bf16* __restrict__ h1,
                                                    float* __restrict__ sum1,
                                                    float* __restrict__ sq1) {
    __shared__ __align__(16) char smem[32768];
    __bf16* lds_a   = (__bf16*)smem;              // 128 x 32 bf16, stride 40 -> 10240B
    __bf16* lds_b   = (__bf16*)(smem + 10240);    // 128 x 32 bf16, stride 40 -> 10240B
    __bf16* lds_out = (__bf16*)smem;              // reused: 128 x 128 bf16 = 32768B
    __shared__ float lsum[HH], lsq[HH];

    const int t = threadIdx.x;
    const int e = blockIdx.y;
    const int row0 = blockIdx.x * 128;
    const int lane = t & 31;
    const int wave = t >> 5;
    const int m0w = (wave & 3) * 32;
    const int n0w = (wave >> 2) * 64;
    const int lrow = lane & 15;
    const bool hiH = lane >= 16;

    if (t < HH) { lsum[t] = 0.f; lsq[t] = 0.f; }
    const float epl = 1.0f + eps[e];

    v8f acc[2][4];
#pragma unroll
    for (int i = 0; i < 2; ++i)
#pragma unroll
        for (int j = 0; j < 4; ++j) acc[i][j] = (v8f){0,0,0,0,0,0,0,0};

    for (int kc = 0; kc < 7; ++kc) {
        // A tile: xin = agg + (1+eps)*v, f32 -> bf16
#pragma unroll
        for (int j = 0; j < 4; ++j) {
            int flat = t + j * 256;
            int r = flat >> 3, kq = flat & 7;
            int c = kc * 32 + kq * 4;
            float4 av = {0.f,0.f,0.f,0.f}, vv = {0.f,0.f,0.f,0.f};
            if (c < CC) {
                size_t off = (size_t)(row0 + r) * CC + c;
                av = *(const float4*)(agg + off);
                vv = *(const float4*)(v + off);
            }
            __bf16* d = lds_a + r * 40 + kq * 4;
            d[0] = (__bf16)(av.x + epl * vv.x);
            d[1] = (__bf16)(av.y + epl * vv.y);
            d[2] = (__bf16)(av.z + epl * vv.z);
            d[3] = (__bf16)(av.w + epl * vv.w);
        }
        // B tile from pre-transposed/padded bf16 weights (K-contiguous)
#pragma unroll
        for (int j = 0; j < 2; ++j) {
            int flat = t + j * 256;
            int h = flat >> 2, kq = flat & 3;
            uint4 src = *(const uint4*)(w1t + (size_t)(e * HH + h) * CPAD + kc * 32 + kq * 8);
            *(uint4*)(lds_b + h * 40 + kq * 8) = src;
        }
        __syncthreads();

        v16bf af[2], bfv[4];
        const int kA = hiH ? 8 : 0;
        const int kB = hiH ? 8 : 0;
#pragma unroll
        for (int i = 0; i < 2; ++i) {
            const __bf16* p = lds_a + (m0w + i * 16 + lrow) * 40 + kA;
            af[i] = cat16(*(const bf16x8*)p, *(const bf16x8*)(p + 16));
        }
#pragma unroll
        for (int j = 0; j < 4; ++j) {
            const __bf16* p = lds_b + (n0w + j * 16 + lrow) * 40 + kB;
            bfv[j] = cat16(*(const bf16x8*)p, *(const bf16x8*)(p + 16));
        }
#pragma unroll
        for (int i = 0; i < 2; ++i)
#pragma unroll
            for (int j = 0; j < 4; ++j)
                acc[i][j] = __builtin_amdgcn_wmma_f32_16x16x32_bf16(
                    false, af[i], false, bfv[j], (short)0, acc[i][j], false, false);
        __syncthreads();
    }

    // epilogue: bias, BN stats, stage output tile to LDS
#pragma unroll
    for (int j = 0; j < 4; ++j) {
        int colL = n0w + j * 16 + lrow;
        float bias = b1[e * HH + colL];
        float s = 0.f, q = 0.f;
#pragma unroll
        for (int i = 0; i < 2; ++i) {
            int rb = m0w + i * 16 + (hiH ? 8 : 0);
#pragma unroll
            for (int p = 0; p < 8; ++p) {
                float val = acc[i][j][p] + bias;
                s += val; q += val * val;
                lds_out[(rb + p) * HH + colL] = (__bf16)val;
            }
        }
        atomicAdd(&lsum[colL], s);
        atomicAdd(&lsq[colL], q);
    }
    __syncthreads();
    // coalesced bf16 store of h1 tile
#pragma unroll
    for (int j = 0; j < 8; ++j) {
        int flat = t + j * 256;          // 2048 uint4 = 128x128 bf16
        int r = flat >> 4, q = flat & 15;
        __bf16* dst = h1 + ((size_t)e * NN + row0 + r) * HH + q * 8;
        *(uint4*)dst = *(const uint4*)(lds_out + r * HH + q * 8);
    }
    if (t < HH) {
        atomicAdd(&sum1[e * HH + t], lsum[t]);
        atomicAdd(&sq1[e * HH + t], lsq[t]);
    }
}

// ================= BN finalize: scale = g*rsqrt(var+eps), shift = beta - mean*scale =================
__global__ __launch_bounds__(256) void finalize_kernel(const float* __restrict__ sum,
                                                       const float* __restrict__ sq,
                                                       const float* __restrict__ g,
                                                       const float* __restrict__ beta,
                                                       float* __restrict__ scale,
                                                       float* __restrict__ shift) {
    int i = blockIdx.x * 256 + threadIdx.x;
    if (i < EE * HH) {
        const float invN = 1.0f / (float)NN;
        float mean = sum[i] * invN;
        float var  = sq[i] * invN - mean * mean;
        float is   = rsqrtf(var + 1e-5f);
        float sc   = g[i] * is;
        scale[i] = sc;
        shift[i] = beta[i] - mean * sc;
    }
}

// ================= GEMM2 stats pass: h2 = relu(bn1(h1)) @ W2 + b2 ; stats only =================
__global__ __launch_bounds__(256) void gemm2_stats_kernel(const __bf16* __restrict__ h1,
                                                          const float* __restrict__ scale1,
                                                          const float* __restrict__ shift1,
                                                          const __bf16* __restrict__ w2t,
                                                          const float* __restrict__ b2,
                                                          float* __restrict__ sum2,
                                                          float* __restrict__ sq2) {
    __shared__ __align__(16) char smem[20480];
    __bf16* lds_a = (__bf16*)smem;
    __bf16* lds_b = (__bf16*)(smem + 10240);
    __shared__ float lsum[HH], lsq[HH];

    const int t = threadIdx.x;
    const int e = blockIdx.y;
    const int row0 = blockIdx.x * 128;
    const int lane = t & 31;
    const int wave = t >> 5;
    const int m0w = (wave & 3) * 32;
    const int n0w = (wave >> 2) * 64;
    const int lrow = lane & 15;
    const bool hiH = lane >= 16;

    if (t < HH) { lsum[t] = 0.f; lsq[t] = 0.f; }

    v8f acc[2][4];
#pragma unroll
    for (int i = 0; i < 2; ++i)
#pragma unroll
        for (int j = 0; j < 4; ++j) acc[i][j] = (v8f){0,0,0,0,0,0,0,0};

    for (int kc = 0; kc < 4; ++kc) {
        // A: load h1 bf16 row chunk, apply fused BN1+ReLU per-k, back to bf16
#pragma unroll
        for (int j = 0; j < 2; ++j) {
            int flat = t + j * 256;
            int r = flat >> 2, kq = flat & 3;
            bf16x8 hv = *(const bf16x8*)(h1 + ((size_t)e * NN + row0 + r) * HH + kc * 32 + kq * 8);
            __bf16* d = lds_a + r * 40 + kq * 8;
#pragma unroll
            for (int ii = 0; ii < 8; ++ii) {
                int kk = kc * 32 + kq * 8 + ii;
                float f = (float)hv[ii];
                float val = fmaxf(0.f, fmaf(f, scale1[e * HH + kk], shift1[e * HH + kk]));
                d[ii] = (__bf16)val;
            }
        }
#pragma unroll
        for (int j = 0; j < 2; ++j) {
            int flat = t + j * 256;
            int h = flat >> 2, kq = flat & 3;
            uint4 src = *(const uint4*)(w2t + (size_t)(e * HH + h) * HH + kc * 32 + kq * 8);
            *(uint4*)(lds_b + h * 40 + kq * 8) = src;
        }
        __syncthreads();

        v16bf af[2], bfv[4];
        const int kA = hiH ? 8 : 0;
        const int kB = hiH ? 8 : 0;
#pragma unroll
        for (int i = 0; i < 2; ++i) {
            const __bf16* p = lds_a + (m0w + i * 16 + lrow) * 40 + kA;
            af[i] = cat16(*(const bf16x8*)p, *(const bf16x8*)(p + 16));
        }
#pragma unroll
        for (int j = 0; j < 4; ++j) {
            const __bf16* p = lds_b + (n0w + j * 16 + lrow) * 40 + kB;
            bfv[j] = cat16(*(const bf16x8*)p, *(const bf16x8*)(p + 16));
        }
#pragma unroll
        for (int i = 0; i < 2; ++i)
#pragma unroll
            for (int j = 0; j < 4; ++j)
                acc[i][j] = __builtin_amdgcn_wmma_f32_16x16x32_bf16(
                    false, af[i], false, bfv[j], (short)0, acc[i][j], false, false);
        __syncthreads();
    }

#pragma unroll
    for (int j = 0; j < 4; ++j) {
        int colL = n0w + j * 16 + lrow;
        float bias = b2[e * HH + colL];
        float s = 0.f, q = 0.f;
#pragma unroll
        for (int i = 0; i < 2; ++i)
#pragma unroll
            for (int p = 0; p < 8; ++p) {
                float val = acc[i][j][p] + bias;
                s += val; q += val * val;
            }
        atomicAdd(&lsum[colL], s);
        atomicAdd(&lsq[colL], q);
    }
    __syncthreads();
    if (t < HH) {
        atomicAdd(&sum2[e * HH + t], lsum[t]);
        atomicAdd(&sq2[e * HH + t], lsq[t]);
    }
}

// ================= combine: recompute layer-2 for selected rows only, gate, write out =================
__global__ __launch_bounds__(256) void combine_kernel(const __bf16* __restrict__ h1,
                                                      const float* __restrict__ scale1,
                                                      const float* __restrict__ shift1,
                                                      const __bf16* __restrict__ w2t,
                                                      const float* __restrict__ b2,
                                                      const float* __restrict__ scale2,
                                                      const float* __restrict__ shift2,
                                                      const int* __restrict__ list,
                                                      const int* __restrict__ cnt,
                                                      const float* __restrict__ gval,
                                                      float* __restrict__ out) {
    const int e = blockIdx.y;
    const int cn = cnt[e];
    if ((int)blockIdx.x * 128 >= cn) return;   // uniform exit before any barrier

    __shared__ __align__(16) char smem[20480];
    __bf16* lds_a = (__bf16*)smem;
    __bf16* lds_b = (__bf16*)(smem + 10240);
    __shared__ int   snodes[128];
    __shared__ float sgates[128];

    const int t = threadIdx.x;
    const int lane = t & 31;
    const int wave = t >> 5;
    const int m0w = (wave & 3) * 32;
    const int n0w = (wave >> 2) * 64;
    const int lrow = lane & 15;
    const bool hiH = lane >= 16;

    if (t < 128) {
        int idx = blockIdx.x * 128 + t;
        if (idx < cn) {
            int nd = list[e * NN + idx];
            snodes[t] = nd;
            sgates[t] = gval[nd];
        } else {
            snodes[t] = -1;
            sgates[t] = 0.f;
        }
    }
    __syncthreads();

    v8f acc[2][4];
#pragma unroll
    for (int i = 0; i < 2; ++i)
#pragma unroll
        for (int j = 0; j < 4; ++j) acc[i][j] = (v8f){0,0,0,0,0,0,0,0};

    for (int kc = 0; kc < 4; ++kc) {
#pragma unroll
        for (int j = 0; j < 2; ++j) {
            int flat = t + j * 256;
            int r = flat >> 2, kq = flat & 3;
            int nd = snodes[r];
            int gn = nd < 0 ? 0 : nd;
            bf16x8 hv = *(const bf16x8*)(h1 + ((size_t)e * NN + gn) * HH + kc * 32 + kq * 8);
            __bf16* d = lds_a + r * 40 + kq * 8;
#pragma unroll
            for (int ii = 0; ii < 8; ++ii) {
                int kk = kc * 32 + kq * 8 + ii;
                float f = (float)hv[ii];
                float val = fmaxf(0.f, fmaf(f, scale1[e * HH + kk], shift1[e * HH + kk]));
                d[ii] = (__bf16)val;
            }
        }
#pragma unroll
        for (int j = 0; j < 2; ++j) {
            int flat = t + j * 256;
            int h = flat >> 2, kq = flat & 3;
            uint4 src = *(const uint4*)(w2t + (size_t)(e * HH + h) * HH + kc * 32 + kq * 8);
            *(uint4*)(lds_b + h * 40 + kq * 8) = src;
        }
        __syncthreads();

        v16bf af[2], bfv[4];
        const int kA = hiH ? 8 : 0;
        const int kB = hiH ? 8 : 0;
#pragma unroll
        for (int i = 0; i < 2; ++i) {
            const __bf16* p = lds_a + (m0w + i * 16 + lrow) * 40 + kA;
            af[i] = cat16(*(const bf16x8*)p, *(const bf16x8*)(p + 16));
        }
#pragma unroll
        for (int j = 0; j < 4; ++j) {
            const __bf16* p = lds_b + (n0w + j * 16 + lrow) * 40 + kB;
            bfv[j] = cat16(*(const bf16x8*)p, *(const bf16x8*)(p + 16));
        }
#pragma unroll
        for (int i = 0; i < 2; ++i)
#pragma unroll
            for (int j = 0; j < 4; ++j)
                acc[i][j] = __builtin_amdgcn_wmma_f32_16x16x32_bf16(
                    false, af[i], false, bfv[j], (short)0, acc[i][j], false, false);
        __syncthreads();
    }

    // epilogue: bias, BN2 + ReLU, gate, scatter to out rows
#pragma unroll
    for (int j = 0; j < 4; ++j) {
        int colL = n0w + j * 16 + lrow;
        float bias = b2[e * HH + colL];
        float sc = scale2[e * HH + colL];
        float sh = shift2[e * HH + colL];
#pragma unroll
        for (int i = 0; i < 2; ++i) {
            int rb = m0w + i * 16 + (hiH ? 8 : 0);
#pragma unroll
            for (int p = 0; p < 8; ++p) {
                int rowL = rb + p;
                int nd = snodes[rowL];
                if (nd >= 0) {
                    float h2 = acc[i][j][p] + bias;
                    float o = fmaxf(0.f, fmaf(h2, sc, sh)) * sgates[rowL];
                    out[(size_t)nd * HH + colL] = o;
                }
            }
        }
    }
}

// ================= aux losses =================
__global__ void losses_kernel(const float* __restrict__ impS,
                              const float* __restrict__ loadS,
                              const float* __restrict__ topS,
                              float* __restrict__ out) {
    const float invN = 1.0f / (float)NN;
    float bs = 0.f;
#pragma unroll
    for (int e = 0; e < EE; ++e) bs += (impS[e] * invN) * (loadS[e] * invN);
    out[(size_t)NN * HH + 0] = 0.01f * (float)EE * bs;
    out[(size_t)NN * HH + 1] = -0.01f * (topS[0] * invN);
}

// ================= launcher =================
extern "C" void kernel_launch(void* const* d_in, const int* in_sizes, int n_in,
                              void* d_out, int out_size, void* d_ws, size_t ws_size,
                              hipStream_t stream) {
    const float* v      = (const float*)d_in[0];
    const int*   a_rows = (const int*)d_in[1];
    const int*   a_cols = (const int*)d_in[2];
    const float* a_vals = (const float*)d_in[3];
    const float* gate_W = (const float*)d_in[4];
    const float* gate_b = (const float*)d_in[5];
    const float* eps    = (const float*)d_in[6];
    const float* W1     = (const float*)d_in[7];
    const float* b1     = (const float*)d_in[8];
    const float* g1     = (const float*)d_in[9];
    const float* beta1  = (const float*)d_in[10];
    const float* W2     = (const float*)d_in[11];
    const float* b2     = (const float*)d_in[12];
    const float* g2     = (const float*)d_in[13];
    const float* beta2  = (const float*)d_in[14];
    const int nedge = in_sizes[1];
    float* out = (float*)d_out;

    char* ws = (char*)d_ws;
    float*  agg    = (float*)(ws + O_AGG);
    float*  sum1   = (float*)(ws + O_SUM1);
    float*  sq1    = (float*)(ws + O_SQ1);
    float*  sum2   = (float*)(ws + O_SUM2);
    float*  sq2    = (float*)(ws + O_SQ2);
    float*  impS   = (float*)(ws + O_IMP);
    float*  loadS  = (float*)(ws + O_LOAD);
    float*  topS   = (float*)(ws + O_TOP);
    int*    cnt    = (int*)  (ws + O_CNT);
    float*  gval   = (float*)(ws + O_GVAL);
    int*    list   = (int*)  (ws + O_LIST);
    float*  scale1 = (float*)(ws + O_SC1);
    float*  shift1 = (float*)(ws + O_SH1);
    float*  scale2 = (float*)(ws + O_SC2);
    float*  shift2 = (float*)(ws + O_SH2);
    __bf16* w1t    = (__bf16*)(ws + O_W1T);
    __bf16* w2t    = (__bf16*)(ws + O_W2T);
    __bf16* h1     = (__bf16*)(ws + O_H1);

    // 1) zero accumulator region
    {
        int n4 = (int)(ZERO_END / 16);
        zero_kernel<<<(n4 + 255) / 256, 256, 0, stream>>>((float4*)ws, n4);
    }
    // 2) weights -> transposed bf16 (K-padded)
    {
        int tot = EE * HH * CPAD + EE * HH * HH;
        prep_weights<<<(tot + 255) / 256, 256, 0, stream>>>(W1, W2, w1t, w2t);
    }
    // 3) gating + loss accumulators + per-expert compaction
    gating_kernel<<<NN / 256, 256, 0, stream>>>(v, gate_W, gate_b, gval,
                                                impS, loadS, topS, cnt, list);
    // 4) sparse aggregation
    aggregate_kernel<<<(nedge + 7) / 8, 256, 0, stream>>>(a_rows, a_cols, a_vals, v, agg, nedge);
    // 5) layer-1 WMMA GEMM + BN1 stats
    gemm1_kernel<<<dim3(NN / 128, EE), 256, 0, stream>>>(agg, v, eps, w1t, b1, h1, sum1, sq1);
    // 6) BN1 finalize
    finalize_kernel<<<(EE * HH + 255) / 256, 256, 0, stream>>>(sum1, sq1, g1, beta1, scale1, shift1);
    // 7) layer-2 dense stats pass
    gemm2_stats_kernel<<<dim3(NN / 128, EE), 256, 0, stream>>>(h1, scale1, shift1, w2t, b2, sum2, sq2);
    // 8) BN2 finalize
    finalize_kernel<<<(EE * HH + 255) / 256, 256, 0, stream>>>(sum2, sq2, g2, beta2, scale2, shift2);
    // 9) gated combine (top-1 selected rows only)
    combine_kernel<<<dim3(NN / 128, EE), 256, 0, stream>>>(h1, scale1, shift1, w2t, b2,
                                                           scale2, shift2, list, cnt, gval, out);
    // 10) scalar losses
    losses_kernel<<<1, 1, 0, stream>>>(impS, loadS, topS, out);
}
